// CAD_13211319403323
// MI455X (gfx1250) — matrix operand order repaired and checked by reading, not compiled, not run on hardware
//
#include <hip/hip_runtime.h>
#include <hip/hip_bf16.h>

// ---------------------------------------------------------------------------
// Problem constants (from reference): B=16, N=4096, D=512, C=2048, K=1, J=0
// out = [loss (1 f32), score (B*N f32)]
// ---------------------------------------------------------------------------
#define B_SZ 16
#define N_SZ 4096
#define D_SZ 512
#define C_SZ 2048
#define M_TOTAL (B_SZ * N_SZ)      // 65536 rows
#define ROWS_PER_WG 128
#define THREADS 256
#define NWAVES (THREADS / 32)
#define NUM_WGS (M_TOTAL / ROWS_PER_WG)   // 512
#define COLS_PER_GROUP 64
#define A_STRIDE (D_SZ + 8)        // bf16 elements, +8 pad -> 1040B row pitch
#define B_STRIDE (D_SZ + 8)
#define LOSS_SCALE (1000.0f / 65536.0f)   // (1/NU) / (B*N*K)

typedef __bf16 bf16_t;
typedef __attribute__((ext_vector_type(16))) __bf16 v16bf;
typedef __attribute__((ext_vector_type(8)))  float  v8f;

union BF16x16 { uint4 q[2]; v16bf v; };

// ---------------------------------------------------------------------------
// Kernel 1: centroids f32 -> bf16 (workspace) + per-centroid squared norms
// ---------------------------------------------------------------------------
__global__ void __launch_bounds__(256)
prep_centroids(const float* __restrict__ cent,
               bf16_t* __restrict__ cbf,
               float* __restrict__ cnorm) {
    const int c = blockIdx.x;          // one centroid row per block
    const int t = threadIdx.x;
    const float* __restrict__ row = cent + (size_t)c * D_SZ;
    bf16_t* __restrict__ out = cbf + (size_t)c * D_SZ;
    float s = 0.0f;
    for (int i = t; i < D_SZ; i += 256) {
        float v = row[i];
        s += v * v;
        out[i] = (bf16_t)v;
    }
    #pragma unroll
    for (int m = 1; m < 32; m <<= 1) s += __shfl_xor(s, m, 32);
    __shared__ float red[8];
    if ((t & 31) == 0) red[t >> 5] = s;
    __syncthreads();
    if (t == 0) {
        float tot = 0.0f;
        #pragma unroll
        for (int i = 0; i < 8; ++i) tot += red[i];
        cnorm[c] = tot;
    }
}

// ---------------------------------------------------------------------------
// Kernel 2: fused (embeds x centroids^T) GEMM + row-min + sqrt + loss partials
//   Each block: 128 rows, sweeps all 2048 columns.
//   A tile (128x512 bf16) staged once in LDS; B tiles (64x512 bf16) streamed
//   through LDS with register double-buffering. bf16 WMMA, f32 accumulate.
// ---------------------------------------------------------------------------
__global__ void __launch_bounds__(THREADS)
fused_min_dist(const float* __restrict__ embeds,
               const bf16_t* __restrict__ cbf,
               const float* __restrict__ cnorm,
               const float* __restrict__ rptr,
               float* __restrict__ score,      // d_out + 1
               float* __restrict__ partial) {  // [NUM_WGS * NWAVES]
    extern __shared__ char smem[];
    bf16_t* lA = (bf16_t*)smem;                                   // 128 * 520
    bf16_t* lB = (bf16_t*)(smem + ROWS_PER_WG * A_STRIDE * 2);    // 64  * 520
    float*  xn = (float*)(smem + ROWS_PER_WG * A_STRIDE * 2
                               + COLS_PER_GROUP * B_STRIDE * 2);  // 128 f32

    const int tid = threadIdx.x;
    const int wg = blockIdx.x;
    const int rowBase = wg * ROWS_PER_WG;

    // ---- stage A: coalesced f32 loads, convert to bf16 in LDS -------------
    const float4* __restrict__ src =
        (const float4*)(embeds + (size_t)rowBase * D_SZ);
    for (int idx = tid; idx < ROWS_PER_WG * D_SZ / 4; idx += THREADS) {
        float4 v = src[idx];
        int e = idx * 4;
        int r = e >> 9;            // / 512
        int k = e & (D_SZ - 1);
        bf16_t* dst = lA + r * A_STRIDE + k;
        dst[0] = (bf16_t)v.x; dst[1] = (bf16_t)v.y;
        dst[2] = (bf16_t)v.z; dst[3] = (bf16_t)v.w;
    }

    // ---- prefetch first B group into registers ----------------------------
    uint4 breg[16];
    {
        const uint4* __restrict__ gb = (const uint4*)cbf;   // cg = 0
        #pragma unroll
        for (int i = 0; i < 16; ++i) breg[i] = gb[tid + i * THREADS];
    }
    __syncthreads();   // A visible

    // ---- per-row |x|^2 from the bf16-rounded values (consistent w/ dot) ---
    if (tid < ROWS_PER_WG) {
        const bf16_t* rp = lA + tid * A_STRIDE;
        float s = 0.0f;
        for (int k = 0; k < D_SZ; ++k) { float v = (float)rp[k]; s += v * v; }
        xn[tid] = s;
    }
    // ---- commit B group 0 to LDS ------------------------------------------
    #pragma unroll
    for (int i = 0; i < 16; ++i) {
        int idx = tid + i * THREADS;
        int r = idx >> 6;                 // 64 uint4 per 512-elem row
        int k = (idx & 63) * 8;
        *(uint4*)(lB + r * B_STRIDE + k) = breg[i];
    }
    __syncthreads();

    const int wv = tid >> 5;
    const int ln = tid & 31;
    const int lrow = ln & 15;
    const int lhalf = ln >> 4;
    const bf16_t* aRow = lA + (wv * 16 + lrow) * A_STRIDE;

    float minv[8];
    #pragma unroll
    for (int i = 0; i < 8; ++i) minv[i] = 3.4e38f;

    for (int cg = 0; cg < C_SZ; cg += COLS_PER_GROUP) {
        const bool hasNext = (cg + COLS_PER_GROUP) < C_SZ;
        if (hasNext) {   // overlap next group's global loads with WMMA
            const uint4* __restrict__ gb =
                (const uint4*)(cbf + (size_t)(cg + COLS_PER_GROUP) * D_SZ);
            #pragma unroll
            for (int i = 0; i < 16; ++i) breg[i] = gb[tid + i * THREADS];
        }

        v8f acc0 = {}, acc1 = {}, acc2 = {}, acc3 = {};
        const bf16_t* bp = lB + lhalf * 16;
        #pragma unroll 4
        for (int kt = 0; kt < D_SZ; kt += 32) {
            // A fragment: lane<16 holds K{kt..kt+15}, lane>=16 holds K{kt+16..kt+31}
            BF16x16 af;
            af.q[0] = *(const uint4*)(aRow + kt + lhalf * 16);
            af.q[1] = *(const uint4*)(aRow + kt + lhalf * 16 + 8);
            // B fragments: lane group selects K half, lane%16 selects column
            BF16x16 b0, b1, b2, b3;
            b0.q[0] = *(const uint4*)(bp + (0 * 16 + lrow) * B_STRIDE + kt);
            b0.q[1] = *(const uint4*)(bp + (0 * 16 + lrow) * B_STRIDE + kt + 8);
            b1.q[0] = *(const uint4*)(bp + (1 * 16 + lrow) * B_STRIDE + kt);
            b1.q[1] = *(const uint4*)(bp + (1 * 16 + lrow) * B_STRIDE + kt + 8);
            b2.q[0] = *(const uint4*)(bp + (2 * 16 + lrow) * B_STRIDE + kt);
            b2.q[1] = *(const uint4*)(bp + (2 * 16 + lrow) * B_STRIDE + kt + 8);
            b3.q[0] = *(const uint4*)(bp + (3 * 16 + lrow) * B_STRIDE + kt);
            b3.q[1] = *(const uint4*)(bp + (3 * 16 + lrow) * B_STRIDE + kt + 8);

            acc0 = __builtin_amdgcn_wmma_f32_16x16x32_bf16(
                false, af.v, false, b0.v, (short)0, acc0, false, false);
            acc1 = __builtin_amdgcn_wmma_f32_16x16x32_bf16(
                false, af.v, false, b1.v, (short)0, acc1, false, false);
            acc2 = __builtin_amdgcn_wmma_f32_16x16x32_bf16(
                false, af.v, false, b2.v, (short)0, acc2, false, false);
            acc3 = __builtin_amdgcn_wmma_f32_16x16x32_bf16(
                false, af.v, false, b3.v, (short)0, acc3, false, false);
        }

        // fold |c|^2 - 2*dot into the running per-lane minimum
        float cn0 = cnorm[cg + 0 * 16 + lrow];
        float cn1 = cnorm[cg + 1 * 16 + lrow];
        float cn2 = cnorm[cg + 2 * 16 + lrow];
        float cn3 = cnorm[cg + 3 * 16 + lrow];
        #pragma unroll
        for (int i = 0; i < 8; ++i) {
            minv[i] = fminf(minv[i], cn0 - 2.0f * acc0[i]);
            minv[i] = fminf(minv[i], cn1 - 2.0f * acc1[i]);
            minv[i] = fminf(minv[i], cn2 - 2.0f * acc2[i]);
            minv[i] = fminf(minv[i], cn3 - 2.0f * acc3[i]);
        }

        if (hasNext) {
            __syncthreads();   // everyone done reading lB
            #pragma unroll
            for (int i = 0; i < 16; ++i) {
                int idx = tid + i * THREADS;
                int r = idx >> 6;
                int k = (idx & 63) * 8;
                *(uint4*)(lB + r * B_STRIDE + k) = breg[i];
            }
            __syncthreads();   // next B tile visible
        }
    }

    // ---- min across the 16 lanes of each half (C/D layout: lane<16 rows 0-7)
    #pragma unroll
    for (int m = 1; m < 16; m <<= 1) {
        #pragma unroll
        for (int i = 0; i < 8; ++i)
            minv[i] = fminf(minv[i], __shfl_xor(minv[i], m, 16));
    }

    const float rv = rptr[0];
    const float r2 = rv * rv;
    float lsum = 0.0f;
    float sc[8];
    #pragma unroll
    for (int i = 0; i < 8; ++i) {
        int localRow = wv * 16 + lhalf * 8 + i;
        float d2 = xn[localRow] + minv[i];
        float d = sqrtf(fmaxf(d2, 0.0f));
        sc[i] = d;
        lsum += fmaxf(d - r2, 0.0f);
    }
    float wsum = lsum + __shfl_xor(lsum, 16, 32);   // both halves

    if (ln == 0 || ln == 16) {
        int base = rowBase + wv * 16 + lhalf * 8;
        #pragma unroll
        for (int i = 0; i < 8; ++i) score[base + i] = sc[i];
    }
    if (ln == 0) partial[wg * NWAVES + wv] = wsum;
}

// ---------------------------------------------------------------------------
// Kernel 3: deterministic tree reduction of loss partials
// ---------------------------------------------------------------------------
__global__ void __launch_bounds__(256)
finalize_loss(const float* __restrict__ partial, float* __restrict__ out) {
    __shared__ float s[256];
    const int t = threadIdx.x;
    float a = 0.0f;
    for (int i = t; i < NUM_WGS * NWAVES; i += 256) a += partial[i];
    s[t] = a;
    __syncthreads();
    for (int k = 128; k > 0; k >>= 1) {
        if (t < k) s[t] += s[t + k];
        __syncthreads();
    }
    if (t == 0) out[0] = s[0] * LOSS_SCALE;
}

// ---------------------------------------------------------------------------
// Launch. d_in: [embeds f32, centroids f32, r f32(1)]. d_out: [loss, score].
// ws layout: [cbf 2MB bf16][cnorm 8KB f32][partials 16KB f32]
// ---------------------------------------------------------------------------
extern "C" void kernel_launch(void* const* d_in, const int* in_sizes, int n_in,
                              void* d_out, int out_size, void* d_ws, size_t ws_size,
                              hipStream_t stream) {
    const float* embeds = (const float*)d_in[0];
    const float* cent   = (const float*)d_in[1];
    const float* r      = (const float*)d_in[2];
    float* out = (float*)d_out;

    bf16_t* cbf   = (bf16_t*)d_ws;
    float* cnorm  = (float*)((char*)d_ws + (size_t)C_SZ * D_SZ * sizeof(bf16_t));
    float* partl  = (float*)((char*)d_ws + (size_t)C_SZ * D_SZ * sizeof(bf16_t)
                                         + (size_t)C_SZ * sizeof(float));

    prep_centroids<<<C_SZ, 256, 0, stream>>>(cent, cbf, cnorm);

    size_t shmem = (size_t)ROWS_PER_WG * A_STRIDE * sizeof(bf16_t)
                 + (size_t)COLS_PER_GROUP * B_STRIDE * sizeof(bf16_t)
                 + (size_t)ROWS_PER_WG * sizeof(float);   // ~200 KB < 320 KB/WGP
    fused_min_dist<<<NUM_WGS, THREADS, shmem, stream>>>(
        embeds, cbf, cnorm, r, out + 1, partl);

    finalize_loss<<<1, 256, 0, stream>>>(partl, out);
}
